// MultiHeadAttention_19963007992315
// MI455X (gfx1250) — matrix operand order repaired and checked by reading, not compiled, not run on hardware
//
#include <hip/hip_runtime.h>
#include <hip/hip_bf16.h>

// MI455X / gfx1250 multi-head attention forward.
// All GEMM-shaped math via V_WMMA_F32_16X16X32_BF16 (wave32, fp32 accum).

typedef __bf16 bf16;
typedef bf16  v16bf __attribute__((ext_vector_type(16)));
typedef bf16  v8bf  __attribute__((ext_vector_type(8)));
typedef float v8f   __attribute__((ext_vector_type(8)));

#define B_    2
#define L_    2048
#define D_    1024
#define H_    16
#define DK_   64
#define M_    (B_*L_)                 // 4096 rows
// (1/sqrt(d_k)) * log2(e): scores scaled into base-2 exponent domain
#define SCALE_LOG2 0.1803368787f

// Load one 16-element bf16 fragment as two b128 chunks: p[0..7], (p+second)[0..7]
static __device__ __forceinline__ v16bf ldfrag(const bf16* p, int second) {
  union { v16bf v; v8bf h[2]; } u;
  u.h[0] = *reinterpret_cast<const v8bf*>(p);
  u.h[1] = *reinterpret_cast<const v8bf*>(p + second);
  return u.v;
}

static __device__ __forceinline__ v8f wmma_bf16(v16bf a, v16bf b, v8f c) {
  // (neg_a, A, neg_b, B, c_mod, C, reuse_a, reuse_b)
  return __builtin_amdgcn_wmma_f32_16x16x32_bf16(false, a, false, b, (short)0, c,
                                                 false, false);
}

static __device__ __forceinline__ float red_max16(float v) {
#pragma unroll
  for (int m = 1; m < 16; m <<= 1) v = fmaxf(v, __shfl_xor(v, m, 32));
  return v;
}
static __device__ __forceinline__ float red_sum16(float v) {
#pragma unroll
  for (int m = 1; m < 16; m <<= 1) v += __shfl_xor(v, m, 32);
  return v;
}

// ---------------------------------------------------------------- convert
__global__ void f32_to_bf16(const float* __restrict__ s, bf16* __restrict__ d,
                            int n) {
  int i  = blockIdx.x * blockDim.x + threadIdx.x;
  int st = gridDim.x * blockDim.x;
  for (; i < n; i += st) d[i] = (bf16)s[i];
}

// ---------------------------------------------------------------- GEMM
// out = A(bf16,[M,1024]) @ W(bf16,[N=1024,K=1024], torch [out,in]).T + bias
// Each wave computes a 16x64 tile: 32 K-steps x 4 WMMAs.
// mode 0: write bf16 [b,h,l,d]   (q, k for attention)
// mode 1: write bf16 [b,h,d,l]   (v transposed -> contiguous-K B frags in PV)
// mode 2: write fp32 [m,n]       (final output)
__global__ void __launch_bounds__(256)
gemm16(const bf16* __restrict__ A, const bf16* __restrict__ W,
       const float* __restrict__ bias, void* __restrict__ outp, int mode) {
  const int lane = threadIdx.x & 31;
  const int wid  = threadIdx.x >> 5;
  const int gw   = blockIdx.x * 8 + wid;   // 4096 waves total
  const int m0   = (gw >> 4) * 16;         // 256 M-tiles
  const int n0   = (gw & 15) * 64;         // 16  N-tiles
  const int half = lane >> 4;
  const int row  = lane & 15;

  v8f acc[4] = {};
  const bf16* ap = A + (size_t)(m0 + row) * D_ + half * 8;
#pragma unroll 4
  for (int k0 = 0; k0 < D_; k0 += 32) {
    v16bf a = ldfrag(ap + k0, 16);  // A: K {h*8..+7} U {h*8+16..+7}
#pragma unroll
    for (int t = 0; t < 4; t++) {
      // B(k,n) = W[n][k]: lane=n row, K {h*8..+7} U {h*8+16..+7}
      v16bf b = ldfrag(W + (size_t)(n0 + t * 16 + row) * D_ + k0 + half * 8, 16);
      acc[t] = wmma_bf16(a, b, acc[t]);
    }
  }
#pragma unroll
  for (int t = 0; t < 4; t++) {
#pragma unroll
    for (int i = 0; i < 8; i++) {
      const int m = m0 + i + 8 * half;       // C layout: VGPR i -> row i+8*half
      const int n = n0 + t * 16 + row;       //           lane&15 -> col
      const float v = acc[t][i] + bias[n];
      if (mode == 2) {
        ((float*)outp)[(size_t)m * D_ + n] = v;
      } else {
        const int b = m >> 11, l = m & 2047; // m = b*2048 + l
        const int h = n >> 6, d = n & 63;    // n = h*64 + d
        bf16* o = (bf16*)outp;
        if (mode == 0)
          o[(((size_t)(b * H_ + h)) * L_ + l) * DK_ + d] = (bf16)v;
        else
          o[(((size_t)(b * H_ + h)) * DK_ + d) * L_ + l] = (bf16)v;
      }
    }
  }
}

// ---------------------------------------------------------------- attention
// One wave = 16 queries of one (b,h). Online (flash) softmax, causal.
// Per 32-key step: 4 score WMMAs + LDS transpose of P + 4 PV WMMAs.
__global__ void __launch_bounds__(128)
flash_attn(const bf16* __restrict__ qh, const bf16* __restrict__ kh,
           const bf16* __restrict__ vT, bf16* __restrict__ attn) {
  __shared__ bf16 smem[4 * 16 * 40];  // per-wave 16x32 P tile, padded stride 40
  const int lane  = threadIdx.x & 31;
  const int wid   = threadIdx.x >> 5;
  const int gw    = blockIdx.x * 4 + wid;  // 4096 waves
  const int bh    = gw >> 7;               // 32 (b,h) pairs
  const int qbase = (gw & 127) * 16;       // 128 query tiles
  const int half  = lane >> 4;
  const int row   = lane & 15;
  bf16* plds = smem + wid * 16 * 40;

  const bf16* qp = qh + (size_t)bh * L_ * DK_;
  const bf16* kp = kh + (size_t)bh * L_ * DK_;
  const bf16* vp = vT + (size_t)bh * DK_ * L_;

  // Q A-fragments (dk = 64 -> two K=32 fragments), kept in registers
  const v16bf a0 = ldfrag(qp + (size_t)(qbase + row) * DK_ + half * 8, 16);
  const v16bf a1 = ldfrag(qp + (size_t)(qbase + row) * DK_ + 32 + half * 8, 16);

  v8f acc[4] = {};
  float mrow[8], lrow[8];
#pragma unroll
  for (int i = 0; i < 8; i++) { mrow[i] = -1e30f; lrow[i] = 0.f; }

  const int nsteps = (qbase + 16 + 31) >> 5;  // causal: keys 0..qbase+15
  for (int s = 0; s < nsteps; ++s) {
    const int kb = s * 32;
    v8f s0 = {}, s1 = {};
    {  // scores: B(kdim, key) = k[key][kdim], lane = key
      const bf16* k0p = kp + (size_t)(kb + row) * DK_ + half * 8;
      s0 = wmma_bf16(a0, ldfrag(k0p, 16), s0);
      s0 = wmma_bf16(a1, ldfrag(k0p + 32, 16), s0);
      const bf16* k1p = kp + (size_t)(kb + 16 + row) * DK_ + half * 8;
      s1 = wmma_bf16(a0, ldfrag(k1p, 16), s1);
      s1 = wmma_bf16(a1, ldfrag(k1p + 32, 16), s1);
    }
#pragma unroll
    for (int i = 0; i < 8; i++) {
      const int qrow = qbase + i + 8 * half;
      float x0 = (kb + row      <= qrow) ? s0[i] * SCALE_LOG2 : -1e30f;
      float x1 = (kb + 16 + row <= qrow) ? s1[i] * SCALE_LOG2 : -1e30f;
      float rm   = red_max16(fmaxf(x0, x1));
      float mnew = fmaxf(mrow[i], rm);
      float p0   = exp2f(x0 - mnew);
      float p1   = exp2f(x1 - mnew);
      float corr = exp2f(mrow[i] - mnew);
      mrow[i] = mnew;
      lrow[i] = lrow[i] * corr + red_sum16(p0 + p1);
#pragma unroll
      for (int t = 0; t < 4; t++) acc[t][i] *= corr;
      const int m = i + 8 * half;          // P tile row-major in LDS
      plds[m * 40 + row]      = (bf16)p0;
      plds[m * 40 + 16 + row] = (bf16)p1;
    }
    // cross-lane LDS RAW inside one wave: DS is in-order, make it explicit
    asm volatile("s_wait_dscnt 0" ::: "memory");
    const v16bf pa = ldfrag(plds + row * 40 + half * 8, 16);  // A frag of P
#pragma unroll
    for (int t = 0; t < 4; t++) {
      // B(kpos, d) = vT[d][kpos]: lane = d, kpos {h*8..+7} U {h*8+16..+7}
      v16bf bv = ldfrag(vp + (size_t)(t * 16 + row) * L_ + kb + half * 8, 16);
      acc[t] = wmma_bf16(pa, bv, acc[t]);
    }
  }
  const int b = bh >> 4, h = bh & 15;
#pragma unroll
  for (int t = 0; t < 4; t++) {
#pragma unroll
    for (int i = 0; i < 8; i++) {
      const int qrow = qbase + i + 8 * half;
      const int d    = t * 16 + row;
      attn[((size_t)b * L_ + qrow) * D_ + h * DK_ + d] =
          (bf16)(acc[t][i] / lrow[i]);
    }
  }
}

// ---------------------------------------------------------------- launch
extern "C" void kernel_launch(void* const* d_in, const int* in_sizes, int n_in,
                              void* d_out, int out_size, void* d_ws,
                              size_t ws_size, hipStream_t stream) {
  (void)in_sizes; (void)n_in; (void)out_size; (void)ws_size;
  const float* Q  = (const float*)d_in[0];
  const float* K  = (const float*)d_in[1];
  const float* V  = (const float*)d_in[2];
  // d_in[3] is the causal mask; applied analytically in flash_attn
  const float* Wq = (const float*)d_in[4];
  const float* bq = (const float*)d_in[5];
  const float* Wk = (const float*)d_in[6];
  const float* bk = (const float*)d_in[7];
  const float* Wv = (const float*)d_in[8];
  const float* bv = (const float*)d_in[9];
  const float* Wo = (const float*)d_in[10];
  const float* bo = (const float*)d_in[11];

  char* ws = (char*)d_ws;
  const size_t SZX = (size_t)M_ * D_ * sizeof(bf16);  // 8 MB activations
  const size_t SZW = (size_t)D_ * D_ * sizeof(bf16);  // 2 MB weights
  bf16* Qb  = (bf16*)(ws);
  bf16* Kb  = (bf16*)(ws + SZX);
  bf16* Vb  = (bf16*)(ws + 2 * SZX);
  bf16* Wqb = (bf16*)(ws + 3 * SZX);
  bf16* Wkb = (bf16*)(ws + 3 * SZX + SZW);
  bf16* Wvb = (bf16*)(ws + 3 * SZX + 2 * SZW);
  bf16* Wob = (bf16*)(ws + 3 * SZX + 3 * SZW);
  bf16* qh  = (bf16*)(ws + 3 * SZX + 4 * SZW);
  bf16* kh  = (bf16*)(ws + 4 * SZX + 4 * SZW);
  bf16* vT  = (bf16*)(ws + 5 * SZX + 4 * SZW);
  bf16* at  = (bf16*)(ws + 6 * SZX + 4 * SZW);  // total 64 MB

  f32_to_bf16<<<512, 256, 0, stream>>>(Q, Qb, M_ * D_);
  f32_to_bf16<<<512, 256, 0, stream>>>(K, Kb, M_ * D_);
  f32_to_bf16<<<512, 256, 0, stream>>>(V, Vb, M_ * D_);
  f32_to_bf16<<<256, 256, 0, stream>>>(Wq, Wqb, D_ * D_);
  f32_to_bf16<<<256, 256, 0, stream>>>(Wk, Wkb, D_ * D_);
  f32_to_bf16<<<256, 256, 0, stream>>>(Wv, Wvb, D_ * D_);
  f32_to_bf16<<<256, 256, 0, stream>>>(Wo, Wob, D_ * D_);

  gemm16<<<512, 256, 0, stream>>>(Qb, Wqb, bq, qh, 0);  // q [b,h,l,d]
  gemm16<<<512, 256, 0, stream>>>(Kb, Wkb, bk, kh, 0);  // k [b,h,l,d]
  gemm16<<<512, 256, 0, stream>>>(Vb, Wvb, bv, vT, 1);  // v [b,h,d,l]

  flash_attn<<<1024, 128, 0, stream>>>(qh, kh, vT, at);

  gemm16<<<512, 256, 0, stream>>>(at, Wob, bo, d_out, 2);  // fp32 out
}
